// Attention_68633577390434
// MI455X (gfx1250) — compile-verified, matrix operand order checked by reading, not run on hardware
//
#include <hip/hip_runtime.h>
#include <hip/hip_bf16.h>

typedef __attribute__((ext_vector_type(16))) _Float16 v16h;
typedef __attribute__((ext_vector_type(8)))  _Float16 v8h;
typedef __attribute__((ext_vector_type(8)))  float    v8f;

#define WMMA_F16(a, b, c) \
    __builtin_amdgcn_wmma_f32_16x16x32_f16(false, (a), false, (b), (short)0, (c), false, false)

constexpr int BB   = 8;     // batch
constexpr int CC   = 512;   // channels
constexpr int NN   = 2048;  // sequence length
constexpr int HH   = 8;     // heads
constexpr int DH   = 64;    // head dim
constexpr int HID  = 512;   // h*d
constexpr int MQKV = 1536;  // 3*hid
constexpr float QK_SCALE = 0.125f; // 64^-0.5

// Load one 16-lane-striped f16 WMMA fragment: caller passes base already offset
// by (row|col)*stride + 8*half; element e<8 from p[e], e>=8 from p[16+e-8],
// matching K(e) = (e&7) + 8*half + 16*(e>>3).
__device__ __forceinline__ v16h load_frag(const _Float16* p) {
    union { v16h v; v8h h[2]; } u;
    u.h[0] = *(const v8h*)(p);
    u.h[1] = *(const v8h*)(p + 16);
    return u.v;
}

// CDNA5 async global->LDS copy, 16B per lane, GV addressing (ASYNCcnt-tracked).
__device__ __forceinline__ void async_copy16(const _Float16* lds_dst, const _Float16* gsrc) {
    unsigned lds_off = (unsigned)(uintptr_t)lds_dst;     // low 32 bits of generic LDS ptr = LDS byte addr
    unsigned long long ga = (unsigned long long)(uintptr_t)gsrc;
    asm volatile("global_load_async_to_lds_b128 %0, %1, off"
                 :: "v"(lds_off), "v"(ga) : "memory");
}
__device__ __forceinline__ void wait_async0() {
    asm volatile("s_wait_asynccnt 0x0" ::: "memory");
}

// ---------------------------------------------------------------- prep kernels
__global__ void cvt_f32_to_f16(const float* __restrict__ src, _Float16* __restrict__ dst, int n) {
    int i = blockIdx.x * blockDim.x + threadIdx.x;
    if (i < n) dst[i] = (_Float16)src[i];
}

// x [B, C, N] f32  ->  xh [B, N, C] f16  (tiled transpose)
__global__ __launch_bounds__(256) void transpose_cvt(const float* __restrict__ x, _Float16* __restrict__ xh) {
    __shared__ float tile[32][33];
    const int b  = blockIdx.z;
    const int n0 = blockIdx.x * 32;
    const int c0 = blockIdx.y * 32;
    const int tx = threadIdx.x, ty = threadIdx.y;
#pragma unroll
    for (int k = 0; k < 4; ++k)
        tile[ty + k * 8][tx] = x[((size_t)b * CC + (c0 + ty + k * 8)) * NN + n0 + tx];
    __syncthreads();
#pragma unroll
    for (int k = 0; k < 4; ++k)
        xh[((size_t)b * NN + (n0 + ty + k * 8)) * CC + c0 + tx] = (_Float16)tile[tx][ty + k * 8];
}

// ---------------------------------------------------------------- shared GEMM core
// Block = 4 waves splitting a 64(M) x 64(N) tile; K = 512.
// B tile (64 cols x 32 k, 4KB) is async-DMA'd into LDS, double-buffered, shared
// by all waves; A fragments are per-wave register-pipelined from global.
constexpr int BSTR = 40;   // B-tile row stride (f16): 80B, 16B-aligned, bank-spread

__device__ __forceinline__ void issue_b_tile(_Float16* bt, const _Float16* bbase,
                                             int n0, int ks, int tid) {
#pragma unroll
    for (int u = 0; u < 2; ++u) {
        const int tr  = tid + u * 128;        // 0..255 -> 64 rows x 4 segs of 16B
        const int row = tr >> 2, seg = tr & 3;
        async_copy16(bt + row * BSTR + seg * 8,
                     bbase + (size_t)(n0 + row) * CC + ks + seg * 8);
    }
}

// ---------------------------------------------------------------- QKV GEMM
// Epilogue scatters into attention-friendly layouts:
//   q,k: [b,h,n,64] f16 (q pre-scaled), v: [b,h,64,n] f16
__global__ __launch_bounds__(128) void qkv_gemm(const _Float16* __restrict__ wq,
                                                const _Float16* __restrict__ xh,
                                                _Float16* __restrict__ qh,
                                                _Float16* __restrict__ kh,
                                                _Float16* __restrict__ vh) {
    __shared__ __align__(16) _Float16 bt[2][64 * BSTR];

    const int tid  = threadIdx.x;
    const int lane = tid & 31, wave = tid >> 5;
    const int half = lane >> 4, idx = lane & 15;
    const int b  = blockIdx.z;
    const int o0 = blockIdx.y * 64 + wave * 16;
    const int n0 = blockIdx.x * 64;

    const _Float16* arow  = wq + (size_t)(o0 + idx) * CC + 8 * half;
    const _Float16* bbase = xh + (size_t)b * NN * CC;

    v8f acc[4];
#pragma unroll
    for (int t = 0; t < 4; ++t)
#pragma unroll
        for (int r = 0; r < 8; ++r) acc[t][r] = 0.0f;

    issue_b_tile(bt[0], bbase, n0, 0, tid);
    v16h a_c = load_frag(arow);
    wait_async0();
    __syncthreads();

    int p = 0;
#pragma unroll
    for (int ks = 0; ks < CC; ks += 32) {
        v16h a_n;
        if (ks + 32 < CC) {                   // prefetch next step
            issue_b_tile(bt[p ^ 1], bbase, n0, ks + 32, tid);
            a_n = load_frag(arow + ks + 32);
        }
#pragma unroll
        for (int t = 0; t < 4; ++t) {
            const v16h bf = load_frag(&bt[p][(t * 16 + idx) * BSTR + 8 * half]);
            acc[t] = WMMA_F16(a_c, bf, acc[t]);
        }
        wait_async0();
        __syncthreads();
        if (ks + 32 < CC) a_c = a_n;
        p ^= 1;
    }

    const int head  = (o0 % HID) / DH;
    const int d0    = o0 % DH;
    const int rbase = 8 * half;
    const size_t bh = (size_t)(b * HH + head);

    if (o0 < 2 * HID) {                       // q or k -> [b,h,n,64]
        const float mul = (o0 < HID) ? QK_SCALE : 1.0f;
        _Float16* dst = (o0 < HID) ? qh : kh;
#pragma unroll
        for (int t = 0; t < 4; ++t) {
            const int n = n0 + t * 16 + idx;
            v8h pk;
#pragma unroll
            for (int r = 0; r < 8; ++r) pk[r] = (_Float16)(acc[t][r] * mul);
            *(v8h*)(dst + (bh * NN + n) * DH + d0 + rbase) = pk;
        }
    } else {                                  // v -> [b,h,64,n]
#pragma unroll
        for (int t = 0; t < 4; ++t) {
            const int n = n0 + t * 16 + idx;
#pragma unroll
            for (int r = 0; r < 8; ++r)
                vh[(bh * DH + (d0 + rbase + r)) * NN + n] = (_Float16)acc[t][r];
        }
    }
}

// ---------------------------------------------------------------- flash attention
// Block = 8 waves x 16 queries for one (b,h). K/V tiles (32 keys) are staged in
// LDS by async DMA, double-buffered: tile j+1 streams in while tile j computes.
__global__ __launch_bounds__(256) void attn_kernel(const _Float16* __restrict__ qh,
                                                   const _Float16* __restrict__ kh,
                                                   const _Float16* __restrict__ vh,
                                                   _Float16* __restrict__ aoh) {
    constexpr int PSTR = 40;   // p-tile row stride (f16): 80B, 16B-aligned, bank-spread
    constexpr int KSTR = 72;   // K-tile row stride:      144B, 16B-aligned, bank-spread
    constexpr int VSTR = 40;   // V-tile row stride:       80B
    __shared__ __align__(16) _Float16 kt[2][32 * KSTR];
    __shared__ __align__(16) _Float16 vt[2][64 * VSTR];
    __shared__ __align__(16) _Float16 pbuf[8][16 * PSTR];

    const int tid  = threadIdx.x;
    const int lane = tid & 31, wave = tid >> 5;
    const int half = lane >> 4, idx = lane & 15;
    const int b = blockIdx.z, h = blockIdx.y;
    const int i0 = blockIdx.x * 128 + wave * 16;
    const size_t bh = (size_t)(b * HH + h);

    const _Float16* qb = qh + bh * NN * DH;
    const _Float16* kb = kh + bh * NN * DH;
    const _Float16* vb = vh + bh * DH * NN;

    // tile-loader mapping (whole block cooperates; 2 async b128 per thread)
    const int krow = tid >> 3, kseg = tid & 7;   // K: 32 rows x 4KB
    const int vrow = tid >> 2, vseg = tid & 3;   // V: 64 rows x 4KB

    const v16h qa0 = load_frag(qb + (size_t)(i0 + idx) * DH + 8 * half);
    const v16h qa1 = load_frag(qb + (size_t)(i0 + idx) * DH + 32 + 8 * half);

    v8f acc[4];
    float mrun[8], lrun[8];
#pragma unroll
    for (int t = 0; t < 4; ++t)
#pragma unroll
        for (int r = 0; r < 8; ++r) acc[t][r] = 0.0f;
#pragma unroll
    for (int r = 0; r < 8; ++r) { mrun[r] = -3.0e38f; lrun[r] = 0.0f; }

    _Float16* pp = pbuf[wave];

    // prologue: fill buffer 0 with keys [0,32)
    async_copy16(&kt[0][krow * KSTR + kseg * 8], kb + (size_t)krow * DH + kseg * 8);
    async_copy16(&vt[0][vrow * VSTR + vseg * 8], vb + (size_t)vrow * NN + vseg * 8);
    wait_async0();
    __syncthreads();

    int p = 0;
    for (int j0 = 0; j0 < NN; j0 += 32) {
        if (j0 + 32 < NN) {                  // async-prefetch next tile into p^1
            const int jn = j0 + 32;
            async_copy16(&kt[p ^ 1][krow * KSTR + kseg * 8], kb + (size_t)(jn + krow) * DH + kseg * 8);
            async_copy16(&vt[p ^ 1][vrow * VSTR + vseg * 8], vb + (size_t)vrow * NN + jn + vseg * 8);
        }

        // ---- scores: s = q^T k for 16 queries x 32 keys (K tile from LDS)
        const v16h kb0 = load_frag(&kt[p][idx * KSTR + 8 * half]);
        const v16h kb1 = load_frag(&kt[p][idx * KSTR + 32 + 8 * half]);
        const v16h kb2 = load_frag(&kt[p][(16 + idx) * KSTR + 8 * half]);
        const v16h kb3 = load_frag(&kt[p][(16 + idx) * KSTR + 32 + 8 * half]);

        v8f z;
#pragma unroll
        for (int r = 0; r < 8; ++r) z[r] = 0.0f;
        v8f s0 = WMMA_F16(qa0, kb0, z);  s0 = WMMA_F16(qa1, kb1, s0);
        v8f s1 = WMMA_F16(qa0, kb2, z);  s1 = WMMA_F16(qa1, kb3, s1);

        // ---- online softmax: each row lives across the 16 lanes of a half
        float p0[8], p1[8], mnew[8], scl[8];
#pragma unroll
        for (int r = 0; r < 8; ++r) {
            float t = fmaxf(s0[r], s1[r]);
            t = fmaxf(t, __shfl_xor(t, 1, 32));
            t = fmaxf(t, __shfl_xor(t, 2, 32));
            t = fmaxf(t, __shfl_xor(t, 4, 32));
            t = fmaxf(t, __shfl_xor(t, 8, 32));
            mnew[r] = fmaxf(mrun[r], t);
            scl[r]  = __expf(mrun[r] - mnew[r]);
            mrun[r] = mnew[r];
        }
#pragma unroll
        for (int r = 0; r < 8; ++r) {
            p0[r] = __expf(s0[r] - mnew[r]);
            p1[r] = __expf(s1[r] - mnew[r]);
            float s = p0[r] + p1[r];
            s += __shfl_xor(s, 1, 32);
            s += __shfl_xor(s, 2, 32);
            s += __shfl_xor(s, 4, 32);
            s += __shfl_xor(s, 8, 32);
            lrun[r] = lrun[r] * scl[r] + s;
        }
#pragma unroll
        for (int t = 0; t < 4; ++t)
#pragma unroll
            for (int r = 0; r < 8; ++r) acc[t][r] *= scl[r];

        // ---- re-fragment P via per-wave LDS tile (C/D layout -> A layout)
#pragma unroll
        for (int r = 0; r < 8; ++r) {
            pp[(r + 8 * half) * PSTR + idx]      = (_Float16)p0[r];
            pp[(r + 8 * half) * PSTR + 16 + idx] = (_Float16)p1[r];
        }
        const v16h pa = load_frag(pp + (size_t)idx * PSTR + 8 * half);

        // ---- acc += P @ V  (V tile from LDS: row d, contiguous local keys)
#pragma unroll
        for (int t = 0; t < 4; ++t) {
            const v16h vf = load_frag(&vt[p][(t * 16 + idx) * VSTR + 8 * half]);
            acc[t] = WMMA_F16(pa, vf, acc[t]);
        }

        wait_async0();       // next tile landed in LDS
        __syncthreads();     // everyone done reading tile p / writing tile p^1
        p ^= 1;
    }

    // ---- normalize + store attention output as [b, n, h*64+d] f16
    float inv[8];
#pragma unroll
    for (int r = 0; r < 8; ++r) inv[r] = 1.0f / lrun[r];
#pragma unroll
    for (int t = 0; t < 4; ++t) {
        const int c = h * DH + t * 16 + idx;
#pragma unroll
        for (int r = 0; r < 8; ++r) {
            const int i = i0 + r + 8 * half;
            aoh[((size_t)b * NN + i) * HID + c] = (_Float16)(acc[t][r] * inv[r]);
        }
    }
}

// ---------------------------------------------------------------- output projection
// y[b,o,n] = sum_c w_out[o,c] * ao[b,n,c] + bias[o]   (f32 out)
__global__ __launch_bounds__(128) void out_gemm(const _Float16* __restrict__ wo,
                                                const _Float16* __restrict__ aoh,
                                                const float* __restrict__ bias,
                                                float* __restrict__ y) {
    __shared__ __align__(16) _Float16 bt[2][64 * BSTR];

    const int tid  = threadIdx.x;
    const int lane = tid & 31, wave = tid >> 5;
    const int half = lane >> 4, idx = lane & 15;
    const int b  = blockIdx.z;
    const int o0 = blockIdx.y * 64 + wave * 16;
    const int n0 = blockIdx.x * 64;

    const _Float16* arow  = wo + (size_t)(o0 + idx) * CC + 8 * half;
    const _Float16* bbase = aoh + (size_t)b * NN * HID;

    v8f acc[4];
#pragma unroll
    for (int t = 0; t < 4; ++t)
#pragma unroll
        for (int r = 0; r < 8; ++r) acc[t][r] = 0.0f;

    issue_b_tile(bt[0], bbase, n0, 0, tid);
    v16h a_c = load_frag(arow);
    wait_async0();
    __syncthreads();

    int p = 0;
#pragma unroll
    for (int ks = 0; ks < HID; ks += 32) {
        v16h a_n;
        if (ks + 32 < HID) {
            issue_b_tile(bt[p ^ 1], bbase, n0, ks + 32, tid);
            a_n = load_frag(arow + ks + 32);
        }
#pragma unroll
        for (int t = 0; t < 4; ++t) {
            const v16h bf = load_frag(&bt[p][(t * 16 + idx) * BSTR + 8 * half]);
            acc[t] = WMMA_F16(a_c, bf, acc[t]);
        }
        wait_async0();
        __syncthreads();
        if (ks + 32 < HID) a_c = a_n;
        p ^= 1;
    }

    const int rbase = 8 * half;
#pragma unroll
    for (int t = 0; t < 4; ++t) {
        const int n = n0 + t * 16 + idx;
#pragma unroll
        for (int r = 0; r < 8; ++r) {
            const int o = o0 + rbase + r;
            y[((size_t)b * CC + o) * NN + n] = acc[t][r] + bias[o];
        }
    }
}

// ---------------------------------------------------------------- launcher
extern "C" void kernel_launch(void* const* d_in, const int* in_sizes, int n_in,
                              void* d_out, int out_size, void* d_ws, size_t ws_size,
                              hipStream_t stream) {
    const float* x     = (const float*)d_in[0];   // [8, 512, 2048]
    const float* w_qkv = (const float*)d_in[1];   // [1536, 512]
    const float* w_out = (const float*)d_in[2];   // [512, 512]
    const float* b_out = (const float*)d_in[3];   // [512]
    float*       y     = (float*)d_out;           // [8, 512, 2048]

    _Float16* xh    = (_Float16*)d_ws;                 // [B, N, C]
    _Float16* wqkvh = xh    + (size_t)BB * NN * CC;    // [1536, 512]
    _Float16* wouth = wqkvh + (size_t)MQKV * CC;       // [512, 512]
    _Float16* qh    = wouth + (size_t)HID * HID;       // [B, H, N, 64]
    _Float16* kh    = qh    + (size_t)BB * HH * NN * DH;
    _Float16* vh    = kh    + (size_t)BB * HH * NN * DH; // [B, H, 64, N]
    _Float16* aoh   = vh    + (size_t)BB * HH * NN * DH; // [B, N, HID]

    cvt_f32_to_f16<<<(MQKV * CC) / 256, 256, 0, stream>>>(w_qkv, wqkvh, MQKV * CC);
    cvt_f32_to_f16<<<(HID * HID) / 256, 256, 0, stream>>>(w_out, wouth, HID * HID);
    transpose_cvt<<<dim3(NN / 32, CC / 32, BB), dim3(32, 8), 0, stream>>>(x, xh);
    qkv_gemm<<<dim3(NN / 64, MQKV / 64, BB), 128, 0, stream>>>(wqkvh, xh, qh, kh, vh);
    attn_kernel<<<dim3(NN / 128, HH, BB), 256, 0, stream>>>(qh, kh, vh, aoh);
    out_gemm<<<dim3(NN / 64, HID / 64, BB), 128, 0, stream>>>(wouth, aoh, b_out, y);
}
